// BiLSTM_CRF_59648505807470
// MI455X (gfx1250) — matrix operand order repaired and checked by reading, not compiled, not run on hardware
//
#include <hip/hip_runtime.h>
#include <hip/hip_bf16.h>

// ---------------------------------------------------------------------------
// BiLSTM-CRF pipeline for gfx1250 (MI455X).
//   Phase 1: f32 -> bf16 conversion of sentence / weights        (VALU)
//   Phase 2: xp = sentence @ w_ih^T + bias  via v_wmma_f32_16x16x32_bf16
//            with double-buffered async global->LDS staging (ASYNCcnt)
//   Phase 3: persistent per-direction LSTM recurrence (L2-resident w_hh)
//   Phase 4: feats = lstm_out @ w_out^T + b_out                  (VALU)
//   Phase 5: Viterbi scan + backtrace (single workgroup)
// ---------------------------------------------------------------------------

typedef __attribute__((ext_vector_type(16))) __bf16 v16bf;
typedef __attribute__((ext_vector_type(8)))  float  v8f;
typedef __attribute__((ext_vector_type(4)))  int    v4i;

#define TSEQ 8192
#define EDIM 1024
#define HDIM 1024
#define HD   512
#define G4   2048        // 4*HD
#define KTAG 34
#define TAG_START 32
#define TAG_STOP  33
#define NEGV -10000.0f

// ---- gfx1250 async global->LDS path (guarded; fallback = load+ds_store) ----
#if defined(__has_builtin)
#if __has_builtin(__builtin_amdgcn_global_load_async_to_lds_b128)
#define USE_ASYNC_LDS 1
#endif
#endif

#if defined(USE_ASYNC_LDS)
#if __has_builtin(__builtin_amdgcn_s_wait_asynccnt)
#define WAIT_ASYNC() __builtin_amdgcn_s_wait_asynccnt(0)
#else
#define WAIT_ASYNC() asm volatile("s_wait_asynccnt 0x0" ::: "memory")
#endif
#else
#define WAIT_ASYNC() do {} while (0)
#endif

typedef __attribute__((address_space(1))) v4i g_v4i;
typedef __attribute__((address_space(3))) v4i l_v4i;

__device__ __forceinline__ void stage16B(const unsigned short* __restrict__ gsrc,
                                         unsigned short* lds_dst) {
#if defined(USE_ASYNC_LDS)
    __builtin_amdgcn_global_load_async_to_lds_b128(
        (g_v4i*)gsrc, (l_v4i*)lds_dst, 0, 0);
#else
    *(uint4*)lds_dst = *(const uint4*)gsrc;
#endif
}

__device__ __forceinline__ unsigned short f2bf(float f) {
    union { float f; unsigned int u; } v; v.f = f;
    unsigned int u = v.u;
    u += 0x7FFFu + ((u >> 16) & 1u);          // round-to-nearest-even
    return (unsigned short)(u >> 16);
}
__device__ __forceinline__ float bfbits(unsigned int lo16) {
    union { unsigned int u; float f; } v; v.u = lo16 << 16; return v.f;
}
__device__ __forceinline__ float sigm(float x) { return 1.0f / (1.0f + expf(-x)); }

// ---------------------------------------------------------------- Phase 1 --
__global__ void cvt_bf16_kernel(const float* __restrict__ src,
                                unsigned short* __restrict__ dst, int n) {
    for (int i = blockIdx.x * blockDim.x + threadIdx.x; i < n;
         i += gridDim.x * blockDim.x)
        dst[i] = f2bf(src[i]);
}

// ---------------------------------------------------------------- Phase 2 --
// xp[dir] = sentence(bf16) @ w_ih[dir]^T(bf16) + (b_ih+b_hh)[dir]
// Block tile 128(M) x 128(N); 8 waves as 2(M) x 4(N); wave tile 64x32 = 4x2
// WMMA 16x16 tiles.  K loop over EDIM in steps of 32 with double-buffered
// async global->LDS staging of the 128x32 A and B K-slabs.
__global__ __launch_bounds__(256)
void xproj_wmma_kernel(const unsigned short* __restrict__ Abf,   // [T,E]
                       const unsigned short* __restrict__ Wf,    // [G4,E]
                       const unsigned short* __restrict__ Wb,
                       const float* __restrict__ bihf, const float* __restrict__ bhhf,
                       const float* __restrict__ bihb, const float* __restrict__ bhhb,
                       float* __restrict__ xpf, float* __restrict__ xpb) {
    const int dir = blockIdx.z;
    const unsigned short* W = dir ? Wb : Wf;
    const float* b1 = dir ? bihb : bihf;
    const float* b2 = dir ? bhhb : bhhf;
    float* xp = dir ? xpb : xpf;

    const int m0   = blockIdx.y * 128;
    const int n0   = blockIdx.x * 128;
    const int wave = threadIdx.x >> 5;
    const int lane = threadIdx.x & 31;
    const int half = lane >> 4;          // lane-half selects K sub-range
    const int r    = lane & 15;          // M row (A) / N col (B,D)
    const int wm   = (wave & 1) * 64;    // 2 waves along M, 4 tiles each
    const int wn   = (wave >> 1) * 32;   // 4 waves along N, 2 tiles each

    // Double-buffered K-slabs: A 128x32 bf16 (8KB), B 128x32 bf16 (8KB)
    __shared__ __attribute__((aligned(16))) unsigned short sA[2][128][32];
    __shared__ __attribute__((aligned(16))) unsigned short sB[2][128][32];

    v8f acc[4][2];
    #pragma unroll
    for (int mt = 0; mt < 4; ++mt)
        #pragma unroll
        for (int nt = 0; nt < 2; ++nt) acc[mt][nt] = (v8f){0,0,0,0,0,0,0,0};

    // stage one 128x32 K-slab of A and B into LDS buffer `buf`
    auto stage = [&](int buf, int kk) {
        for (int c = threadIdx.x; c < 512; c += 256) {
            const int row = c >> 2;
            const int seg = (c & 3) * 8;          // 8 bf16 = 16 bytes
            stage16B(Abf + (size_t)(m0 + row) * EDIM + kk + seg, &sA[buf][row][seg]);
            stage16B(W   + (size_t)(n0 + row) * EDIM + kk + seg, &sB[buf][row][seg]);
        }
    };

    stage(0, 0);
    for (int kk = 0; kk < EDIM; kk += 32) {
        const int cur = (kk >> 5) & 1;
        WAIT_ASYNC();                 // own async copies into buf[cur] done
        __syncthreads();              // everyone's copies visible; prev reads done
        if (kk + 32 < EDIM) stage(cur ^ 1, kk + 32);

        // --- A fragments (16x32 bf16): lane holds row wm+mt*16+r,
        //     K chunks [half*8, +8) and [16+half*8, +8) (per ISA layout)
        v16bf afr[4];
        #pragma unroll
        for (int mt = 0; mt < 4; ++mt) {
            const unsigned short* ap = &sA[cur][wm + mt * 16 + r][half * 8];
            union { uint4 u[2]; v16bf v; } pk;
            pk.u[0] = *(const uint4*)(ap);        // K = half*8 .. +7
            pk.u[1] = *(const uint4*)(ap + 16);   // K = 16+half*8 .. +7
            afr[mt] = pk.v;
        }
        // --- B fragments (32x16 bf16): lane holds column wn+nt*16+r,
        //     K chunks [half*8, +8) and [16+half*8, +8) (same as A)
        v16bf bfr[2];
        #pragma unroll
        for (int nt = 0; nt < 2; ++nt) {
            const unsigned short* bp = &sB[cur][wn + nt * 16 + r][half * 8];
            union { uint4 u[2]; v16bf v; } pk;
            pk.u[0] = *(const uint4*)(bp);
            pk.u[1] = *(const uint4*)(bp + 16);
            bfr[nt] = pk.v;
        }
        #pragma unroll
        for (int mt = 0; mt < 4; ++mt)
            #pragma unroll
            for (int nt = 0; nt < 2; ++nt)
                acc[mt][nt] = __builtin_amdgcn_wmma_f32_16x16x32_bf16(
                    false, afr[mt], false, bfr[nt],
                    (short)0, acc[mt][nt], false, false);
    }

    // D layout: VGPR j -> M = j + 8*half, N = r
    #pragma unroll
    for (int mt = 0; mt < 4; ++mt) {
        #pragma unroll
        for (int nt = 0; nt < 2; ++nt) {
            const int col = n0 + wn + nt * 16 + r;
            const float bias = b1[col] + b2[col];
            #pragma unroll
            for (int j = 0; j < 8; ++j) {
                const int row = m0 + wm + mt * 16 + j + half * 8;
                xp[(size_t)row * G4 + col] = acc[mt][nt][j] + bias;
            }
        }
    }
}

// ---------------------------------------------------------------- Phase 3 --
// One persistent 1024-thread workgroup per direction. Thread j (<512) owns
// hidden element j; threads j and j+512 split the K=512 dot products of the
// 4 gate rows {j, 512+j, 1024+j, 1536+j}; partials combined through LDS.
__global__ __launch_bounds__(1024)
void lstm_recurrence_kernel(const unsigned short* __restrict__ Whf,  // [G4,HD]
                            const unsigned short* __restrict__ Whb,
                            const float* __restrict__ xpf,           // [T,G4]
                            const float* __restrict__ xpb,
                            const float* __restrict__ h0,            // [2,HD]
                            const float* __restrict__ c0,
                            float* __restrict__ lstm_out) {          // [T,H]
    const int dir = blockIdx.x;
    const unsigned short* W = dir ? Whb : Whf;
    const float* xp = dir ? xpb : xpf;

    __shared__ float sh[HD];
    __shared__ float part[HD][4];

    const int tid = threadIdx.x;
    const int j   = tid & (HD - 1);
    const int kh  = tid >> 9;            // K-half: 0 or 1
    float c = 0.0f;
    if (kh == 0) { sh[j] = h0[dir * HD + j]; c = c0[dir * HD + j]; }
    __syncthreads();

    const int kbeg = kh * 256;
    const unsigned short* w0 = W + (size_t)(j)          * HD + kbeg;
    const unsigned short* w1 = W + (size_t)(HD + j)     * HD + kbeg;
    const unsigned short* w2 = W + (size_t)(2 * HD + j) * HD + kbeg;
    const unsigned short* w3 = W + (size_t)(3 * HD + j) * HD + kbeg;

    for (int step = 0; step < TSEQ; ++step) {
        const int t = dir ? (TSEQ - 1 - step) : step;
        if (step + 1 < TSEQ) {
            const int tn = dir ? (TSEQ - 2 - step) : (step + 1);
            __builtin_prefetch(xp + (size_t)tn * G4 + j, 0, 1);
        }
        float s0 = 0.f, s1 = 0.f, s2 = 0.f, s3 = 0.f;
        #pragma unroll 4
        for (int k = 0; k < 256; k += 2) {
            const float hv0 = sh[kbeg + k];
            const float hv1 = sh[kbeg + k + 1];
            const unsigned int p0 = *(const unsigned int*)(w0 + k);
            const unsigned int p1 = *(const unsigned int*)(w1 + k);
            const unsigned int p2 = *(const unsigned int*)(w2 + k);
            const unsigned int p3 = *(const unsigned int*)(w3 + k);
            s0 = fmaf(bfbits(p0 & 0xffffu), hv0, s0);
            s0 = fmaf(bfbits(p0 >> 16),     hv1, s0);
            s1 = fmaf(bfbits(p1 & 0xffffu), hv0, s1);
            s1 = fmaf(bfbits(p1 >> 16),     hv1, s1);
            s2 = fmaf(bfbits(p2 & 0xffffu), hv0, s2);
            s2 = fmaf(bfbits(p2 >> 16),     hv1, s2);
            s3 = fmaf(bfbits(p3 & 0xffffu), hv0, s3);
            s3 = fmaf(bfbits(p3 >> 16),     hv1, s3);
        }
        if (kh) { part[j][0] = s0; part[j][1] = s1; part[j][2] = s2; part[j][3] = s3; }
        __syncthreads();                 // partials visible, old sh fully read
        if (kh == 0) {
            const float* xr = xp + (size_t)t * G4;
            float gi = s0 + part[j][0] + xr[j];
            float gf = s1 + part[j][1] + xr[HD + j];
            float gg = s2 + part[j][2] + xr[2 * HD + j];
            float go = s3 + part[j][3] + xr[3 * HD + j];
            gi = sigm(gi); gf = sigm(gf); go = sigm(go); gg = tanhf(gg);
            c = gf * c + gi * gg;
            const float h = go * tanhf(c);
            sh[j] = h;
            lstm_out[(size_t)t * HDIM + dir * HD + j] = h;
        }
        __syncthreads();                 // new sh visible to everyone
    }
}

// ---------------------------------------------------------------- Phase 4 --
__global__ __launch_bounds__(64)
void feats_kernel(const float* __restrict__ lstm_out,
                  const float* __restrict__ w_out,
                  const float* __restrict__ b_out,
                  float* __restrict__ feats) {
    const int t = blockIdx.x;
    const int k = threadIdx.x;
    if (k >= KTAG) return;
    const float* x = lstm_out + (size_t)t * HDIM;
    const float* w = w_out + (size_t)k * HDIM;
    float s = b_out[k];
    #pragma unroll 4
    for (int e = 0; e < HDIM; ++e) s = fmaf(x[e], w[e], s);
    feats[t * KTAG + k] = s;
}

// ---------------------------------------------------------------- Phase 5 --
__global__ __launch_bounds__(64)
void viterbi_kernel(const float* __restrict__ feats,
                    const float* __restrict__ trans,
                    int* __restrict__ bps,
                    float* __restrict__ out) {
    __shared__ float tr[KTAG * KTAG];
    __shared__ float fv[KTAG];
    __shared__ float nv[KTAG];
    const int tid = threadIdx.x;
    for (int i = tid; i < KTAG * KTAG; i += 64) tr[i] = trans[i];
    if (tid < KTAG) fv[tid] = (tid == TAG_START) ? 0.0f : NEGV;
    __syncthreads();

    for (int t = 0; t < TSEQ; ++t) {
        if (tid < KTAG) {
            float best = -3.4e38f; int bp = 0;
            #pragma unroll
            for (int p = 0; p < KTAG; ++p) {
                const float s = fv[p] + tr[tid * KTAG + p];
                if (s > best) { best = s; bp = p; }   // first max (argmax)
            }
            nv[tid] = best + feats[t * KTAG + tid];
            bps[t * KTAG + tid] = bp;
        }
        __syncthreads();
        if (tid < KTAG) fv[tid] = nv[tid];
        __syncthreads();
    }

    if (tid == 0) {
        float best = -3.4e38f; int bl = 0;
        for (int k = 0; k < KTAG; ++k) {
            const float s = fv[k] + tr[TAG_STOP * KTAG + k];
            if (s > best) { best = s; bl = k; }
        }
        out[0] = best;                       // path_score
        int tag = bl;
        for (int t = TSEQ - 1; t >= 0; --t) {
            out[1 + t] = (float)tag;         // best_path (reverse scan emit)
            tag = bps[t * KTAG + tag];
        }
    }
}

// ---------------------------------------------------------------------------
extern "C" void kernel_launch(void* const* d_in, const int* in_sizes, int n_in,
                              void* d_out, int out_size, void* d_ws, size_t ws_size,
                              hipStream_t stream) {
    const float* sentence = (const float*)d_in[0];
    const float* h0       = (const float*)d_in[1];
    const float* c0       = (const float*)d_in[2];
    const float* w_ih_f   = (const float*)d_in[3];
    const float* w_hh_f   = (const float*)d_in[4];
    const float* b_ih_f   = (const float*)d_in[5];
    const float* b_hh_f   = (const float*)d_in[6];
    const float* w_ih_b   = (const float*)d_in[7];
    const float* w_hh_b   = (const float*)d_in[8];
    const float* b_ih_b   = (const float*)d_in[9];
    const float* b_hh_b   = (const float*)d_in[10];
    const float* w_out    = (const float*)d_in[11];
    const float* b_out    = (const float*)d_in[12];
    const float* trans    = (const float*)d_in[13];
    float* out = (float*)d_out;

    // ---- workspace carve-up (all offsets 256B-aligned) ----
    char* ws = (char*)d_ws;
    size_t off = 0;
    auto take = [&](size_t bytes) { void* p = ws + off; off += (bytes + 255) & ~size_t(255); return p; };
    unsigned short* sent_bf = (unsigned short*)take((size_t)TSEQ * EDIM * 2);
    unsigned short* wihf_bf = (unsigned short*)take((size_t)G4 * EDIM * 2);
    unsigned short* wihb_bf = (unsigned short*)take((size_t)G4 * EDIM * 2);
    unsigned short* whhf_bf = (unsigned short*)take((size_t)G4 * HD * 2);
    unsigned short* whhb_bf = (unsigned short*)take((size_t)G4 * HD * 2);
    float* xpf      = (float*)take((size_t)TSEQ * G4 * 4);
    float* xpb      = (float*)take((size_t)TSEQ * G4 * 4);
    float* lstm_out = (float*)take((size_t)TSEQ * HDIM * 4);
    float* feats    = (float*)take((size_t)TSEQ * KTAG * 4);
    int*   bps      = (int*)  take((size_t)TSEQ * KTAG * 4);

    // Phase 1: bf16 conversions
    cvt_bf16_kernel<<<2048, 256, 0, stream>>>(sentence, sent_bf, TSEQ * EDIM);
    cvt_bf16_kernel<<<1024, 256, 0, stream>>>(w_ih_f, wihf_bf, G4 * EDIM);
    cvt_bf16_kernel<<<1024, 256, 0, stream>>>(w_ih_b, wihb_bf, G4 * EDIM);
    cvt_bf16_kernel<<< 512, 256, 0, stream>>>(w_hh_f, whhf_bf, G4 * HD);
    cvt_bf16_kernel<<< 512, 256, 0, stream>>>(w_hh_b, whhb_bf, G4 * HD);

    // Phase 2: WMMA input projections (both directions via grid.z)
    dim3 ggrid(G4 / 128, TSEQ / 128, 2);
    xproj_wmma_kernel<<<ggrid, 256, 0, stream>>>(
        sent_bf, wihf_bf, wihb_bf, b_ih_f, b_hh_f, b_ih_b, b_hh_b, xpf, xpb);

    // Phase 3: recurrence, one persistent workgroup per direction
    lstm_recurrence_kernel<<<2, 1024, 0, stream>>>(
        whhf_bf, whhb_bf, xpf, xpb, h0, c0, lstm_out);

    // Phase 4: output projection
    feats_kernel<<<TSEQ, 64, 0, stream>>>(lstm_out, w_out, b_out, feats);

    // Phase 5: Viterbi + backtrace -> out[0] = score, out[1..T] = path
    viterbi_kernel<<<1, 64, 0, stream>>>(feats, trans, bps, out);

    // Third output: the sentence passthrough
    (void)hipMemcpyAsync(out + 1 + TSEQ, sentence,
                         (size_t)TSEQ * EDIM * sizeof(float),
                         hipMemcpyDeviceToDevice, stream);
}
